// SAGE_23055384445756
// MI455X (gfx1250) — compile-verified, matrix operand order checked by reading, not run on hardware
//
#include <hip/hip_runtime.h>
#include <hip/hip_bf16.h>
#include <stdint.h>

#define N_NODES 50000
#define N_EDGES 800000
#define D_FEAT  256
#define N_CLS   47
#define N_CLS_PAD 48
#define K_CAT   512   // [mean | x] concatenated along K

typedef __attribute__((ext_vector_type(8)))  float  v8f;
typedef __attribute__((ext_vector_type(8)))  __bf16 v8bf;
typedef __attribute__((ext_vector_type(16))) __bf16 v16bf;

__device__ __forceinline__ unsigned short f32_to_bf16_rne(float f) {
    union { float f; unsigned int u; } v; v.f = f;
    unsigned int u = v.u;
    unsigned int r = u + 0x7FFFu + ((u >> 16) & 1u);   // round-to-nearest-even
    return (unsigned short)(r >> 16);
}

// ---------------------------------------------------------------- utilities
__global__ void zero_f32(float* __restrict__ p, long long n) {
    long long i = (long long)blockIdx.x * blockDim.x + threadIdx.x;
    long long stride = (long long)gridDim.x * blockDim.x;
    for (; i < n; i += stride) p[i] = 0.0f;
}

__global__ void degree_kernel(const int* __restrict__ dst,
                              float* __restrict__ deg, int E) {
    int e = blockIdx.x * blockDim.x + threadIdx.x;
    if (e < E) atomicAdd(&deg[dst[e]], 1.0f);
}

// Pack [Wl | Wr] (each [dout,256] f32 row-major) into bf16 [dout_pad, 512]
__global__ void pack_weights(const float* __restrict__ Wl,
                             const float* __restrict__ Wr,
                             unsigned short* __restrict__ Wcat,
                             int dout, int dout_pad) {
    int idx = blockIdx.x * blockDim.x + threadIdx.x;
    int total = dout_pad * K_CAT;
    if (idx >= total) return;
    int n = idx >> 9;          // row
    int k = idx & (K_CAT - 1); // col
    float v = 0.0f;
    if (n < dout) v = (k < D_FEAT) ? Wl[n * D_FEAT + k]
                                   : Wr[n * D_FEAT + (k - D_FEAT)];
    Wcat[idx] = f32_to_bf16_rne(v);
}

// ------------------------------------------------------------- aggregation
// One thread handles 4 consecutive floats of one edge message:
// float4 gather from h[src], 4x global_atomic_add_f32 into agg[dst] (L2-resident).
__global__ __launch_bounds__(256)
void scatter_kernel(const float* __restrict__ h,
                    const int* __restrict__ src,
                    const int* __restrict__ dst,
                    float* __restrict__ agg, int E) {
    long long idx = (long long)blockIdx.x * blockDim.x + threadIdx.x;
    long long total = (long long)E * (D_FEAT / 4);
    if (idx >= total) return;
    int e  = (int)(idx >> 6);
    int c4 = ((int)idx & 63) << 2;
    int s = src[e], d = dst[e];
    const float4 val = *(const float4*)(h + (long long)s * D_FEAT + c4);
    float* a = agg + (long long)d * D_FEAT + c4;
    atomicAdd(a + 0, val.x);
    atomicAdd(a + 1, val.y);
    atomicAdd(a + 2, val.z);
    atomicAdd(a + 3, val.w);
}

// mean = agg/max(deg,1); A = bf16([mean | h]) [N, 512]; re-zero agg for next layer
__global__ __launch_bounds__(256)
void prepare_A(float* __restrict__ agg, const float* __restrict__ deg,
               const float* __restrict__ h,
               unsigned short* __restrict__ Abf, int N) {
    long long idx = (long long)blockIdx.x * blockDim.x + threadIdx.x;
    long long total = (long long)N * D_FEAT;
    if (idx >= total) return;
    int i = (int)(idx >> 8);
    int k = (int)idx & (D_FEAT - 1);
    float dg = deg[i];
    dg = dg > 1.0f ? dg : 1.0f;
    float mean = agg[idx] / dg;
    agg[idx] = 0.0f;  // ready for next layer's scatter
    long long row = (long long)i * K_CAT;
    Abf[row + k]          = f32_to_bf16_rne(mean);
    Abf[row + D_FEAT + k] = f32_to_bf16_rne(h[idx]);
}

// ------------------------------------------------------------------- GEMM
// out[M, ldout] = A[M,512](bf16) @ B[NoutPad,512](bf16)^T + bias, optional ReLU.
// One wave computes a 16 x (NT*16) tile: the A fragment for each K-step is
// loaded once and reused by NT consecutive v_wmma_f32_16x16x32_bf16 ops
// (A L2 traffic / NT; loads per wmma drops 4 -> 2.5 at NT=4).
// ISA 16-bit A layout: lanes 0-15 hold row M=lane with K in {kb..kb+7, kb+16..kb+23},
// kb = 0 (lanes<16) or 8 (lanes>=16). B (N-major = W row-major) mirrors this.
template <int NT>
__global__ __launch_bounds__(256)
void sage_wmma_gemm(const __bf16* __restrict__ A,
                    const __bf16* __restrict__ B,
                    const float* __restrict__ bias,
                    float* __restrict__ out,
                    int M, int Nout, int ngroups, int ldout, int relu) {
    int wave = (int)((blockIdx.x * blockDim.x + threadIdx.x) >> 5);
    int lane = threadIdx.x & 31;
    int tiles = (M >> 4) * ngroups;
    if (wave >= tiles) return;
    int mt = wave / ngroups, g = wave - mt * ngroups;
    int m0 = mt << 4;
    int n0 = g * (NT * 16);

    int r  = lane & 15;
    int kb = (lane >> 4) << 3;       // 0 or 8

    const __bf16* pa = A + (long long)(m0 + r) * K_CAT + kb;
    const __bf16* pb = B + (long long)(n0 + r) * K_CAT + kb;

    v8f acc[NT];
#pragma unroll
    for (int t = 0; t < NT; ++t) acc[t] = (v8f){};

#pragma unroll
    for (int kk = 0; kk < K_CAT / 32; ++kk) {
        v8bf alo = *(const v8bf*)(pa + kk * 32);
        v8bf ahi = *(const v8bf*)(pa + kk * 32 + 16);
        v16bf a = __builtin_shufflevector(alo, ahi, 0,1,2,3,4,5,6,7,
                                                     8,9,10,11,12,13,14,15);
#pragma unroll
        for (int t = 0; t < NT; ++t) {
            const __bf16* pbt = pb + (long long)t * 16 * K_CAT;
            v8bf blo = *(const v8bf*)(pbt + kk * 32);
            v8bf bhi = *(const v8bf*)(pbt + kk * 32 + 16);
            v16bf b = __builtin_shufflevector(blo, bhi, 0,1,2,3,4,5,6,7,
                                                         8,9,10,11,12,13,14,15);
            acc[t] = __builtin_amdgcn_wmma_f32_16x16x32_bf16(
                         false, a, false, b, (short)0, acc[t], false, false);
        }
    }

    // C/D layout: lane = column (lanes 0-15 -> rows 0-7, lanes 16-31 -> rows 8-15)
    int rowbase = m0 + ((lane >> 4) << 3);
#pragma unroll
    for (int t = 0; t < NT; ++t) {
        int col = n0 + t * 16 + (lane & 15);
        if (col < Nout) {
            float bv = bias[col];
#pragma unroll
            for (int v = 0; v < 8; ++v) {
                float val = acc[t][v] + bv;
                if (relu) val = val > 0.0f ? val : 0.0f;
                out[(long long)(rowbase + v) * ldout + col] = val;
            }
        }
    }
}

// ------------------------------------------------------------- log_softmax
__global__ void log_softmax_kernel(float* __restrict__ out, int N, int C) {
    int i = blockIdx.x * blockDim.x + threadIdx.x;
    if (i >= N) return;
    float* row = out + (long long)i * C;
    float m = -INFINITY;
    for (int c = 0; c < C; ++c) m = fmaxf(m, row[c]);
    float s = 0.0f;
    for (int c = 0; c < C; ++c) s += expf(row[c] - m);
    float ls = logf(s);
    for (int c = 0; c < C; ++c) row[c] = row[c] - m - ls;
}

// ---------------------------------------------------------------- launcher
extern "C" void kernel_launch(void* const* d_in, const int* in_sizes, int n_in,
                              void* d_out, int out_size, void* d_ws, size_t ws_size,
                              hipStream_t stream) {
    const float* x    = (const float*)d_in[0];
    const int*   edges= (const int*)d_in[1];
    const float* Wl0  = (const float*)d_in[2];
    const float* bl0  = (const float*)d_in[3];
    const float* Wr0  = (const float*)d_in[4];
    const float* Wl1  = (const float*)d_in[5];
    const float* bl1  = (const float*)d_in[6];
    const float* Wr1  = (const float*)d_in[7];
    const float* Wl2  = (const float*)d_in[8];
    const float* bl2  = (const float*)d_in[9];
    const float* Wr2  = (const float*)d_in[10];
    const int* src = edges;            // edge_index[0]
    const int* dst = edges + N_EDGES;  // edge_index[1]
    float* out = (float*)d_out;        // [N, 47]

    // workspace carving (256B aligned)
    char* ws = (char*)d_ws;
    size_t off = 0;
    auto carve = [&](size_t bytes) { char* p = ws + off;
                                     off += (bytes + 255) & ~(size_t)255; return p; };
    float* deg = (float*)carve((size_t)N_NODES * 4);
    float* agg = (float*)carve((size_t)N_NODES * D_FEAT * 4);
    unsigned short* Abf = (unsigned short*)carve((size_t)N_NODES * K_CAT * 2);
    float* hA  = (float*)carve((size_t)N_NODES * D_FEAT * 4);
    float* hB  = (float*)carve((size_t)N_NODES * D_FEAT * 4);
    unsigned short* W0 = (unsigned short*)carve((size_t)D_FEAT * K_CAT * 2);
    unsigned short* W1 = (unsigned short*)carve((size_t)D_FEAT * K_CAT * 2);
    unsigned short* W2 = (unsigned short*)carve((size_t)N_CLS_PAD * K_CAT * 2);
    (void)ws_size; (void)n_in; (void)in_sizes; (void)out_size;

    const int T = 256;

    // init: zero deg + agg, build degrees, pack weights
    zero_f32<<<2048, T, 0, stream>>>(deg, (long long)N_NODES);
    zero_f32<<<4096, T, 0, stream>>>(agg, (long long)N_NODES * D_FEAT);
    degree_kernel<<<(N_EDGES + T - 1) / T, T, 0, stream>>>(dst, deg, N_EDGES);
    pack_weights<<<(D_FEAT * K_CAT + T - 1) / T, T, 0, stream>>>(Wl0, Wr0, W0, D_FEAT, D_FEAT);
    pack_weights<<<(D_FEAT * K_CAT + T - 1) / T, T, 0, stream>>>(Wl1, Wr1, W1, D_FEAT, D_FEAT);
    pack_weights<<<(N_CLS_PAD * K_CAT + T - 1) / T, T, 0, stream>>>(Wl2, Wr2, W2, N_CLS, N_CLS_PAD);

    const long long scatterThreads = (long long)N_EDGES * (D_FEAT / 4);
    const int scatterBlocks = (int)((scatterThreads + T - 1) / T);
    const long long prepThreads = (long long)N_NODES * D_FEAT;
    const int prepBlocks = (int)((prepThreads + T - 1) / T);

    // hidden layers: NoutPad=256, NT=4 -> 4 column groups of 64
    auto gemm256 = [&](const unsigned short* W, const float* bias, float* o) {
        const int ngroups = D_FEAT / 64;                 // 4
        int tiles = (N_NODES / 16) * ngroups;
        int blocks = (tiles * 32 + T - 1) / T;
        sage_wmma_gemm<4><<<blocks, T, 0, stream>>>(
            (const __bf16*)Abf, (const __bf16*)W, bias, o,
            N_NODES, D_FEAT, ngroups, D_FEAT, 1);
    };

    // layer 0: x -> hA
    scatter_kernel<<<scatterBlocks, T, 0, stream>>>(x, src, dst, agg, N_EDGES);
    prepare_A<<<prepBlocks, T, 0, stream>>>(agg, deg, x, Abf, N_NODES);
    gemm256(W0, bl0, hA);

    // layer 1: hA -> hB
    scatter_kernel<<<scatterBlocks, T, 0, stream>>>(hA, src, dst, agg, N_EDGES);
    prepare_A<<<prepBlocks, T, 0, stream>>>(agg, deg, hA, Abf, N_NODES);
    gemm256(W1, bl1, hB);

    // layer 2: hB -> logits in d_out [N,47]; NoutPad=48, NT=3, one column group
    scatter_kernel<<<scatterBlocks, T, 0, stream>>>(hB, src, dst, agg, N_EDGES);
    prepare_A<<<prepBlocks, T, 0, stream>>>(agg, deg, hB, Abf, N_NODES);
    {
        int tiles = (N_NODES / 16) * 1;
        int blocks = (tiles * 32 + T - 1) / T;
        sage_wmma_gemm<3><<<blocks, T, 0, stream>>>(
            (const __bf16*)Abf, (const __bf16*)W2, bl2, out,
            N_NODES, N_CLS, 1, N_CLS, 0);
    }

    // in-place log_softmax over 47 classes
    log_softmax_kernel<<<(N_NODES + T - 1) / T, T, 0, stream>>>(out, N_NODES, N_CLS);
}
